// AttentionBlock_39393440039169
// MI455X (gfx1250) — matrix operand order repaired and checked by reading, not compiled, not run on hardware
//
#include <hip/hip_runtime.h>
#include <hip/hip_bf16.h>

// ---------------------------------------------------------------------------
// AttentionBlock for MI455X (gfx1250): bf16 WMMA pipeline, f32 accumulate.
// GEMM operands laid out so every fragment load is a contiguous 16B
// global_load_b128 (192MB L2 holds the whole working set). GEMM k-loops are
// software-pipelined (register double-buffer) so s_wait_loadcnt no longer
// drains to zero before each v_wmma.
// ---------------------------------------------------------------------------

typedef __bf16 bf16_t;
typedef bf16_t bf16x16 __attribute__((ext_vector_type(16)));
typedef float  f32x8   __attribute__((ext_vector_type(8)));

#define Bn 16
#define Cc 512
#define Nn 1024     // H*W
#define NH 4
#define HD 128
#define NG 8

union FragU { bf16x16 f; uint4 q[2]; };
union AccU  { f32x8 v; float f[8]; };

__device__ __forceinline__ unsigned short f2bf(float x) {
  unsigned int u = __float_as_uint(x);
  return (unsigned short)((u + 0x7FFFu + ((u >> 16) & 1u)) >> 16);
}

// A-matrix 16x32 bf16 fragment from row-major [M][K] (ld in elements).
// lane m holds row m; K = {8h..8h+7, 16+8h..16+8h+7}  -> two 16B loads.
__device__ __forceinline__ bf16x16 load_a_frag(const unsigned short* p, int ld) {
  int lane = threadIdx.x & 31, h = lane >> 4, m = lane & 15;
  const unsigned short* r = p + (size_t)m * ld + 8 * h;
  FragU u;
  u.q[0] = *reinterpret_cast<const uint4*>(r);
  u.q[1] = *reinterpret_cast<const uint4*>(r + 16);
  return u.f;
}

// B-matrix 32x16 bf16 fragment; logical [K][N] stored COLUMN-major as
// S[N][K] row-major (ld in elements). lane n holds col n, K = {8h..8h+7, 16+8h..16+8h+7}.
__device__ __forceinline__ bf16x16 load_b_frag(const unsigned short* p, int ld) {
  int lane = threadIdx.x & 31, h = lane >> 4, n = lane & 15;
  const unsigned short* r = p + (size_t)n * ld + 8 * h;
  FragU u;
  u.q[0] = *reinterpret_cast<const uint4*>(r);
  u.q[1] = *reinterpret_cast<const uint4*>(r + 16);
  return u.f;
}

__device__ __forceinline__ f32x8 wmma_bf16(bf16x16 a, bf16x16 b, f32x8 c) {
  return __builtin_amdgcn_wmma_f32_16x16x32_bf16(false, a, false, b,
                                                 (short)0, c, false, false);
}

// ---------------------------------------------------------------- kernel 0
__global__ void k_conv_weights(const float* __restrict__ wqkv,
                               const float* __restrict__ wproj,
                               unsigned short* __restrict__ wqb,
                               unsigned short* __restrict__ wpb) {
  int i = blockIdx.x * 256 + threadIdx.x;          // 4096 blocks -> 1048576
  if (i < 3 * Cc * Cc) wqb[i] = f2bf(wqkv[i]);
  else                 wpb[i - 3 * Cc * Cc] = f2bf(wproj[i - 3 * Cc * Cc]);
}

// ---------------------------------------------------------------- kernel 1
// One block per (b,g): contiguous 64ch*1024 = 65536-float reduction.
__global__ void k_gn_stats(const float* __restrict__ x,
                           float* __restrict__ mean, float* __restrict__ rstd) {
  int bg = blockIdx.x;
  const float4* p = reinterpret_cast<const float4*>(x + (size_t)bg * 65536);
  float s = 0.f, s2 = 0.f;
  for (int i = threadIdx.x; i < 16384; i += 256) {
    float4 v = p[i];
    s  += v.x + v.y + v.z + v.w;
    s2 += v.x * v.x + v.y * v.y + v.z * v.z + v.w * v.w;
  }
  __shared__ float r1[256], r2[256];
  r1[threadIdx.x] = s; r2[threadIdx.x] = s2;
  __syncthreads();
  for (int off = 128; off > 0; off >>= 1) {
    if ((int)threadIdx.x < off) {
      r1[threadIdx.x] += r1[threadIdx.x + off];
      r2[threadIdx.x] += r2[threadIdx.x + off];
    }
    __syncthreads();
  }
  if (threadIdx.x == 0) {
    float mu  = r1[0] * (1.f / 65536.f);
    float var = r2[0] * (1.f / 65536.f) - mu * mu;
    mean[bg] = mu;
    rstd[bg] = rsqrtf(var + 1e-5f);
  }
}

// ---------------------------------------------------------------- kernel 2
// Normalize + transpose [C][N] -> xnT[b][n][c] bf16 via 64x64 LDS tile.
__global__ void k_gn_apply(const float* __restrict__ x,
                           const float* __restrict__ gamma,
                           const float* __restrict__ beta,
                           const float* __restrict__ mean,
                           const float* __restrict__ rstd,
                           unsigned short* __restrict__ xnT) {
  int b = blockIdx.z, c0 = blockIdx.y * 64, n0 = blockIdx.x * 64;
  __shared__ unsigned short tile[64][72];
#pragma unroll
  for (int q = 0; q < 4; q++) {
    int idx = threadIdx.x + q * 256;       // 0..1023
    int cl = idx >> 4, nl = (idx & 15) * 4;
    int c = c0 + cl, g = c >> 6;
    float mu = mean[b * NG + g], rs = rstd[b * NG + g];
    float ga = gamma[c], be = beta[c];
    float4 v = *reinterpret_cast<const float4*>(
        x + ((size_t)b * Cc + c) * Nn + n0 + nl);
    tile[cl][nl + 0] = f2bf((v.x - mu) * rs * ga + be);
    tile[cl][nl + 1] = f2bf((v.y - mu) * rs * ga + be);
    tile[cl][nl + 2] = f2bf((v.z - mu) * rs * ga + be);
    tile[cl][nl + 3] = f2bf((v.w - mu) * rs * ga + be);
  }
  __syncthreads();
#pragma unroll
  for (int q = 0; q < 4; q++) {
    int idx = threadIdx.x + q * 256;
    int nl = idx >> 4, ce = (idx & 15) * 4;
    union { unsigned short s[4]; uint2 u; } pk;
#pragma unroll
    for (int e = 0; e < 4; e++) pk.s[e] = tile[ce + e][nl];
    *reinterpret_cast<uint2*>(xnT + ((size_t)b * Nn + n0 + nl) * Cc + c0 + ce) = pk.u;
  }
}

// ---------------------------------------------------------------- kernel 3
// QKV GEMM per batch: D[1536][1024] = Wqkv(bf16,[d][c]) x Xnorm([c][n]).
// 4 waves/block, each wave: 32d x 64n (8 WMMAs/k-step), K pipelined.
__global__ void k_qkv(const unsigned short* __restrict__ wqb,
                      const unsigned short* __restrict__ xnT,
                      unsigned short* __restrict__ qT,
                      unsigned short* __restrict__ kT,
                      unsigned short* __restrict__ vN) {
  int b = blockIdx.z;
  int n0 = blockIdx.x * 64;
  int wave = threadIdx.x >> 5;
  int d_wave = blockIdx.y * 128 + wave * 32;

  const unsigned short* abase = wqb + (size_t)d_wave * Cc;
  const unsigned short* bbase = xnT + ((size_t)b * Nn + n0) * Cc;

  f32x8 acc[2][4] = {};
  bf16x16 a0 = load_a_frag(abase, Cc);
  bf16x16 a1 = load_a_frag(abase + (size_t)16 * Cc, Cc);
  bf16x16 bfr[4];
#pragma unroll
  for (int t = 0; t < 4; t++)
    bfr[t] = load_b_frag(bbase + (size_t)t * 16 * Cc, Cc);

#pragma unroll 4
  for (int k0 = 0; k0 < Cc; k0 += 32) {
    int kn = (k0 + 32) & (Cc - 1);                 // wrapped prefetch
    bf16x16 a0n = load_a_frag(abase + kn, Cc);
    bf16x16 a1n = load_a_frag(abase + (size_t)16 * Cc + kn, Cc);
    bf16x16 bn[4];
#pragma unroll
    for (int t = 0; t < 4; t++)
      bn[t] = load_b_frag(bbase + (size_t)t * 16 * Cc + kn, Cc);
#pragma unroll
    for (int t = 0; t < 4; t++) {
      acc[0][t] = wmma_bf16(a0, bfr[t], acc[0][t]);
      acc[1][t] = wmma_bf16(a1, bfr[t], acc[1][t]);
    }
    a0 = a0n; a1 = a1n;
#pragma unroll
    for (int t = 0; t < 4; t++) bfr[t] = bn[t];
  }

  int lane = threadIdx.x & 31, h = lane >> 4, col = lane & 15;
#pragma unroll
  for (int dsub = 0; dsub < 2; dsub++) {
    int d_eff = d_wave + dsub * 16;
#pragma unroll
    for (int t = 0; t < 4; t++) {
      AccU u; u.v = acc[dsub][t];
      int nidx = n0 + t * 16 + col;
      if (d_eff < Cc) {                                 // Q -> transposed
        int head = d_eff >> 7, cbase = d_eff & 127;
        union { unsigned short s[8]; uint4 q; } pk;
#pragma unroll
        for (int r = 0; r < 8; r++) pk.s[r] = f2bf(u.f[r]);
        *reinterpret_cast<uint4*>(
            qT + (((size_t)b * NH + head) * Nn + nidx) * HD + cbase + 8 * h) = pk.q;
      } else if (d_eff < 2 * Cc) {                      // K -> transposed
        int dd = d_eff - Cc, head = dd >> 7, cbase = dd & 127;
        union { unsigned short s[8]; uint4 q; } pk;
#pragma unroll
        for (int r = 0; r < 8; r++) pk.s[r] = f2bf(u.f[r]);
        *reinterpret_cast<uint4*>(
            kT + (((size_t)b * NH + head) * Nn + nidx) * HD + cbase + 8 * h) = pk.q;
      } else {                                          // V -> natural [c][n]
        int dd = d_eff - 2 * Cc, head = dd >> 7, cbase = dd & 127;
#pragma unroll
        for (int r = 0; r < 8; r++)
          vN[(((size_t)b * NH + head) * HD + cbase + r + 8 * h) * Nn + nidx] =
              f2bf(u.f[r]);
      }
    }
  }
}

// ---------------------------------------------------------------- kernel 4
// Flash attention: one wave per 16-row query tile, stream j in steps of 32,
// online softmax. V fragments are issued before the softmax so their latency
// overlaps the transcendental-heavy VALU work. LDS P-tile is wave-private
// (no block barriers; within-wave DS ordering is enforced by dscnt waits).
__global__ void k_attn(const unsigned short* __restrict__ qT,
                       const unsigned short* __restrict__ kT,
                       const unsigned short* __restrict__ vN,
                       unsigned short* __restrict__ outT) {
  int wave = threadIdx.x >> 5;
  int wt = blockIdx.x * 4 + wave;
  int b = wt >> 8, rem = wt & 255, head = rem >> 6, i0 = (rem & 63) * 16;
  int lane = threadIdx.x & 31, h = lane >> 4, col = lane & 15;

  const float scale = 0.08838834764831845f;  // 1/sqrt(128)
  const size_t bh = (size_t)b * NH + head;

  bf16x16 aq[4];
  const unsigned short* qb = qT + (bh * Nn + i0) * HD;
#pragma unroll
  for (int kk = 0; kk < 4; kk++) aq[kk] = load_a_frag(qb + 32 * kk, HD);

  f32x8 o[8] = {};
  float mrow[8], lrow[8];
#pragma unroll
  for (int r = 0; r < 8; r++) { mrow[r] = -1e30f; lrow[r] = 0.f; }

  __shared__ unsigned short pl[4][16 * 32];
  unsigned short* myp = pl[wave];

  const unsigned short* kb0 = kT + bh * Nn * HD;
  const unsigned short* vb0 = vN + bh * HD * Nn;

  for (int j0 = 0; j0 < Nn; j0 += 32) {
    const unsigned short* kb = kb0 + (size_t)j0 * HD;
    f32x8 s0 = {}, s1 = {};
#pragma unroll
    for (int kk = 0; kk < 4; kk++) {
      bf16x16 bk0 = load_b_frag(kb + kk * 32, HD);
      bf16x16 bk1 = load_b_frag(kb + (size_t)16 * HD + kk * 32, HD);
      s0 = wmma_bf16(aq[kk], bk0, s0);
      s1 = wmma_bf16(aq[kk], bk1, s1);
    }

    // V fragments in flight during the softmax.
    bf16x16 bv[8];
#pragma unroll
    for (int cc = 0; cc < 8; cc++)
      bv[cc] = load_b_frag(vb0 + (size_t)cc * 16 * Nn + j0, Nn);

    AccU u0, u1; u0.v = s0; u1.v = s1;
    float alpha[8];
#pragma unroll
    for (int r = 0; r < 8; r++) {
      float x0 = u0.f[r] * scale, x1 = u1.f[r] * scale;
      float vm = fmaxf(x0, x1);
      vm = fmaxf(vm, __shfl_xor(vm, 8, 32));
      vm = fmaxf(vm, __shfl_xor(vm, 4, 32));
      vm = fmaxf(vm, __shfl_xor(vm, 2, 32));
      vm = fmaxf(vm, __shfl_xor(vm, 1, 32));
      float mn = fmaxf(mrow[r], vm);
      float a  = __expf(mrow[r] - mn);
      float p0 = __expf(x0 - mn), p1 = __expf(x1 - mn);
      float rs = p0 + p1;
      rs += __shfl_xor(rs, 8, 32);
      rs += __shfl_xor(rs, 4, 32);
      rs += __shfl_xor(rs, 2, 32);
      rs += __shfl_xor(rs, 1, 32);
      lrow[r] = lrow[r] * a + rs;
      mrow[r] = mn;
      alpha[r] = a;
      myp[(r + 8 * h) * 32 + col]      = f2bf(p0);
      myp[(r + 8 * h) * 32 + 16 + col] = f2bf(p1);
    }
#pragma unroll
    for (int cc = 0; cc < 8; cc++) {
      AccU uo; uo.v = o[cc];
#pragma unroll
      for (int r = 0; r < 8; r++) uo.f[r] *= alpha[r];
      o[cc] = uo.v;
    }
    bf16x16 pa = load_a_frag(myp, 32);   // ds reads ordered after ds writes
#pragma unroll
    for (int cc = 0; cc < 8; cc++) o[cc] = wmma_bf16(pa, bv[cc], o[cc]);
  }

  unsigned short* ob = outT + (bh * Nn + i0) * HD;
#pragma unroll
  for (int cc = 0; cc < 8; cc++) {
    AccU uo; uo.v = o[cc];
#pragma unroll
    for (int r = 0; r < 8; r++)
      ob[(size_t)(r + 8 * h) * HD + cc * 16 + col] =
          f2bf(uo.f[r] * (1.0f / lrow[r]));
  }
}

// ---------------------------------------------------------------- kernel 5
// Output projection + bias + residual. Wave: 32i x 64d, K pipelined.
__global__ void k_proj(const unsigned short* __restrict__ outT,
                       const unsigned short* __restrict__ wpb,
                       const float* __restrict__ bproj,
                       const float* __restrict__ x,
                       float* __restrict__ out) {
  int b = blockIdx.z;
  int d0 = blockIdx.x * 64;
  int wave = threadIdx.x >> 5;
  int i_wave = blockIdx.y * 128 + wave * 32;

  const unsigned short* abase = outT + (size_t)b * NH * Nn * HD;

  auto a_ptr = [&](int k0, int isub) {
    int hh = k0 >> 7, coff = k0 & 127;
    return abase + ((size_t)hh * Nn + i_wave + 16 * isub) * HD + coff;
  };

  f32x8 acc[2][4] = {};
  bf16x16 a0 = load_a_frag(a_ptr(0, 0), HD);
  bf16x16 a1 = load_a_frag(a_ptr(0, 1), HD);
  bf16x16 bfr[4];
#pragma unroll
  for (int t = 0; t < 4; t++)
    bfr[t] = load_b_frag(wpb + (size_t)(d0 + t * 16) * Cc, Cc);

#pragma unroll 4
  for (int k0 = 0; k0 < Cc; k0 += 32) {
    int kn = (k0 + 32) & (Cc - 1);
    bf16x16 a0n = load_a_frag(a_ptr(kn, 0), HD);
    bf16x16 a1n = load_a_frag(a_ptr(kn, 1), HD);
    bf16x16 bn[4];
#pragma unroll
    for (int t = 0; t < 4; t++)
      bn[t] = load_b_frag(wpb + (size_t)(d0 + t * 16) * Cc + kn, Cc);
#pragma unroll
    for (int t = 0; t < 4; t++) {
      acc[0][t] = wmma_bf16(a0, bfr[t], acc[0][t]);
      acc[1][t] = wmma_bf16(a1, bfr[t], acc[1][t]);
    }
    a0 = a0n; a1 = a1n;
#pragma unroll
    for (int t = 0; t < 4; t++) bfr[t] = bn[t];
  }

  int lane = threadIdx.x & 31, h = lane >> 4, col = lane & 15;
#pragma unroll
  for (int isub = 0; isub < 2; isub++) {
#pragma unroll
    for (int t = 0; t < 4; t++) {
      int d = d0 + t * 16 + col;
      float bias = bproj[d];
      size_t off = ((size_t)b * Cc + d) * Nn + i_wave + 16 * isub + 8 * h;
      float4 x0 = *reinterpret_cast<const float4*>(x + off);
      float4 x1 = *reinterpret_cast<const float4*>(x + off + 4);
      AccU u; u.v = acc[isub][t];
      float4 r0 = make_float4(x0.x + bias + u.f[0], x0.y + bias + u.f[1],
                              x0.z + bias + u.f[2], x0.w + bias + u.f[3]);
      float4 r1 = make_float4(x1.x + bias + u.f[4], x1.y + bias + u.f[5],
                              x1.z + bias + u.f[6], x1.w + bias + u.f[7]);
      *reinterpret_cast<float4*>(out + off)     = r0;
      *reinterpret_cast<float4*>(out + off + 4) = r1;
    }
  }
}

// ---------------------------------------------------------------------------
extern "C" void kernel_launch(void* const* d_in, const int* in_sizes, int n_in,
                              void* d_out, int out_size, void* d_ws,
                              size_t ws_size, hipStream_t stream) {
  const float* x     = (const float*)d_in[0];
  const float* gamma = (const float*)d_in[1];
  const float* beta  = (const float*)d_in[2];
  const float* wqkv  = (const float*)d_in[3];
  const float* wproj = (const float*)d_in[4];
  const float* bproj = (const float*)d_in[5];
  float* out = (float*)d_out;

  // Workspace carve-up (~82 MB total).
  char* w = (char*)d_ws;
  float* meanp = (float*)w;            w += 512;
  float* rstdp = (float*)w;            w += 512;
  unsigned short* xnT  = (unsigned short*)w; w += (size_t)Bn * Nn * Cc * 2;
  unsigned short* wqb  = (unsigned short*)w; w += (size_t)3 * Cc * Cc * 2;
  unsigned short* wpb  = (unsigned short*)w; w += (size_t)Cc * Cc * 2;
  unsigned short* qT   = (unsigned short*)w; w += (size_t)Bn * NH * Nn * HD * 2;
  unsigned short* kTp  = (unsigned short*)w; w += (size_t)Bn * NH * Nn * HD * 2;
  unsigned short* vNp  = (unsigned short*)w; w += (size_t)Bn * NH * HD * Nn * 2;
  unsigned short* oT   = (unsigned short*)w;

  k_conv_weights<<<4096, 256, 0, stream>>>(wqkv, wproj, wqb, wpb);
  k_gn_stats<<<Bn * NG, 256, 0, stream>>>(x, meanp, rstdp);
  k_gn_apply<<<dim3(Nn / 64, Cc / 64, Bn), 256, 0, stream>>>(
      x, gamma, beta, meanp, rstdp, xnT);
  k_qkv<<<dim3(Nn / 64, (3 * Cc) / 128, Bn), 128, 0, stream>>>(
      wqb, xnT, qT, kTp, vNp);
  k_attn<<<(Bn * NH * (Nn / 16)) / 4, 128, 0, stream>>>(qT, kTp, vNp, oT);
  k_proj<<<dim3(Cc / 64, Nn / 128, Bn), 128, 0, stream>>>(
      oT, wpb, bproj, x, out);
}
